// Basic_GATv2_80874234184377
// MI455X (gfx1250) — matrix operand order repaired and checked by reading, not compiled, not run on hardware
//
#include <hip/hip_runtime.h>

typedef __attribute__((ext_vector_type(16))) __bf16 v16bf;
typedef __attribute__((ext_vector_type(8)))  __bf16 v8bf;
typedef __attribute__((ext_vector_type(8)))  float  v8f;

#define B_   4
#define N_   512
#define F_   128
#define OUT_ 128
#define H_   8

#define USE_ASYNC_LDS 1

__device__ __forceinline__ unsigned short f2bf(float x){
  __bf16 h = (__bf16)x;
  return __builtin_bit_cast(unsigned short, h);
}
__device__ __forceinline__ unsigned int pk2(float a, float b){
  return (unsigned int)f2bf(a) | ((unsigned int)f2bf(b) << 16);
}
__device__ __forceinline__ v16bf cat8(v8bf lo, v8bf hi){
  return __builtin_shufflevector(lo, hi, 0,1,2,3,4,5,6,7,8,9,10,11,12,13,14,15);
}
__device__ __forceinline__ v8bf cvt8(float4 a, float4 b){
  v8bf r;
  r[0] = (__bf16)a.x; r[1] = (__bf16)a.y; r[2] = (__bf16)a.z; r[3] = (__bf16)a.w;
  r[4] = (__bf16)b.x; r[5] = (__bf16)b.y; r[6] = (__bf16)b.z; r[7] = (__bf16)b.w;
  return r;
}
__device__ __forceinline__ v16bf ldfrag(const unsigned short* q){
  return cat8(*(const v8bf*)q, *(const v8bf*)(q + 8));
}

// ---------------------------------------------------------------------------
// Kernel W: swizzle 5 weight matrices (128x128 f32) into bf16 B-fragment
// layout [nt][kb][lane][e]: k = kb*32 + (e>>3)*16 + (lane>>4)*8 + (e&7), n = nt*16+(lane&15);
// compute gbias[b][n] = graph@wg_w + (w1_b+w2_b+we_b+wg_b).
// ---------------------------------------------------------------------------
__global__ void prep_kernel(const float* __restrict__ m_w, const float* __restrict__ w1_w,
                            const float* __restrict__ w2_w, const float* __restrict__ skip_w,
                            const float* __restrict__ we_w, const float* __restrict__ graph,
                            const float* __restrict__ w1_b, const float* __restrict__ w2_b,
                            const float* __restrict__ we_b, const float* __restrict__ wg_b,
                            const float* __restrict__ wg_w,
                            unsigned short* __restrict__ wsw, float* __restrict__ gbias){
  int idx = blockIdx.x * 256 + threadIdx.x;
  if (idx < 5*16384){
    int w  = idx >> 14;
    int r  = idx & 16383;
    int e  = r & 15;
    int l  = (r >> 4) & 31;
    int kb = (r >> 9) & 3;
    int nt = (r >> 11) & 7;
    int k = kb*32 + (e>>3)*16 + (l>>4)*8 + (e&7);
    int n = nt*16 + (l & 15);
    const float* W = (w==0)?m_w:(w==1)?w1_w:(w==2)?w2_w:(w==3)?skip_w:we_w;
    wsw[idx] = f2bf(W[k*128 + n]);
  } else if (idx < 5*16384 + 512){
    int g = idx - 5*16384;
    int b = g >> 7, n = g & 127;
    float acc = w1_b[n] + w2_b[n] + we_b[n] + wg_b[n];
    for (int k = 0; k < 128; ++k) acc += graph[b*128 + k] * wg_w[k*128 + n];
    gbias[g] = acc;
  }
}

// ---------------------------------------------------------------------------
// Kernel A: per-node projections via WMMA (one wave per weight matrix).
// B fragments are software-pipelined (nt+1 loads issue before nt's WMMAs).
// pre1 is stored TRANSPOSED: pre1T[b][n][j]  (n = output column, j = node).
// ---------------------------------------------------------------------------
__global__ __launch_bounds__(128) void proj_kernel(
    const float* __restrict__ node, const unsigned short* __restrict__ wsw,
    const float* __restrict__ gbias, const float* __restrict__ m_b,
    const float* __restrict__ skip_b,
    float* __restrict__ vals, float* __restrict__ pre1T,
    float* __restrict__ rowbias, float* __restrict__ skipout){
  __shared__ __align__(16) unsigned short nbuf[2048];   // 16 x 128 bf16
  int tid = threadIdx.x;
  int b = blockIdx.x >> 5;
  int nbase = (blockIdx.x & 31) * 16;
  for (int it = 0; it < 4; ++it){
    int f4  = it*128 + tid;
    int row = f4 >> 5;
    int c4  = f4 & 31;
    float4 v = *(const float4*)(node + ((size_t)(b*N_ + nbase + row)*F_ + c4*4));
    *(uint2*)&nbuf[row*128 + c4*4] = make_uint2(pk2(v.x, v.y), pk2(v.z, v.w));
  }
  __syncthreads();
  int lane = tid & 31;
  int wvs = __builtin_amdgcn_readfirstlane(tid >> 5);   // uniform wave id
  int rowm = lane & 15, sel = (lane >> 4) * 8;
  v16bf afr[4];
  for (int kb = 0; kb < 4; ++kb){
    const unsigned short* p = &nbuf[rowm*128 + kb*32 + sel];
    afr[kb] = cat8(*(const v8bf*)p, *(const v8bf*)(p + 16));
  }
  const unsigned short* wb = wsw + wvs*16384 + lane*16;
  int nl = lane & 15;
  // pipelined B fragments
  v16bf bcur[4], bnxt[4];
#pragma unroll
  for (int kb = 0; kb < 4; ++kb) bcur[kb] = ldfrag(wb + kb*512);
  for (int nt = 0; nt < 8; ++nt){
    if (nt < 7){
#pragma unroll
      for (int kb = 0; kb < 4; ++kb) bnxt[kb] = ldfrag(wb + ((nt+1)*4 + kb)*512);
    }
    v8f acc = {};
#pragma unroll
    for (int kb = 0; kb < 4; ++kb){
      acc = __builtin_amdgcn_wmma_f32_16x16x32_bf16(false, afr[kb], false, bcur[kb],
                                                    (short)0, acc, false, false);
    }
    int col = nt*16 + nl;
    if (wvs == 0){
      float bias = m_b[col];
#pragma unroll
      for (int r = 0; r < 8; ++r){
        int row = nbase + r + 8*(lane >> 4);
        vals[(((size_t)b*H_ + nt)*N_ + row)*16 + nl] = acc[r] + bias;
      }
    } else if (wvs == 1){
#pragma unroll
      for (int r = 0; r < 8; ++r){
        int row = nbase + r + 8*(lane >> 4);
        pre1T[((size_t)(b*OUT_ + col))*N_ + row] = acc[r];     // transposed
      }
    } else if (wvs == 2){
      float bias = gbias[b*128 + col];
#pragma unroll
      for (int r = 0; r < 8; ++r){
        int row = nbase + r + 8*(lane >> 4);
        rowbias[((size_t)(b*N_ + row))*OUT_ + col] = acc[r] + bias;
      }
    } else {
      float bias = skip_b[col];
#pragma unroll
      for (int r = 0; r < 8; ++r){
        int row = nbase + r + 8*(lane >> 4);
        skipout[((size_t)(b*N_ + row))*OUT_ + col] = acc[r] + bias;
      }
    }
#pragma unroll
    for (int kb = 0; kb < 4; ++kb) bcur[kb] = bnxt[kb];
  }
}

// ---------------------------------------------------------------------------
// Staging: one 16x128 f32 edge tile is a contiguous 8KB range; each of 256
// threads DMAs two float4's to LDS via the CDNA5 async path (ASYNCcnt).
// ---------------------------------------------------------------------------
__device__ __forceinline__ void stage_tile(const float* __restrict__ gsrc,
                                           float* lds_base, int tid){
#if USE_ASYNC_LDS
  {
    unsigned l0 = (unsigned)(size_t)(lds_base + tid*4);
    const float* g0 = gsrc + tid*4;
    asm volatile("global_load_async_to_lds_b128 %0, %1, off"
                 :: "v"(l0), "v"(g0) : "memory");
    unsigned l1 = (unsigned)(size_t)(lds_base + (tid+256)*4);
    const float* g1 = gsrc + (tid+256)*4;
    asm volatile("global_load_async_to_lds_b128 %0, %1, off"
                 :: "v"(l1), "v"(g1) : "memory");
  }
#else
  float4 a = *(const float4*)(gsrc + tid*4);
  float4 b = *(const float4*)(gsrc + (tid+256)*4);
  *(float4*)(lds_base + tid*4) = a;
  *(float4*)(lds_base + (tid+256)*4) = b;
#endif
}
__device__ __forceinline__ void stage_fence(){
#if USE_ASYNC_LDS
  asm volatile("s_wait_asynccnt 0x0" ::: "memory");
#endif
}

// ---------------------------------------------------------------------------
// Kernel B: fused edge-GEMM + logits + softmax + attention + skip + LN.
// grid = B*N blocks (one (b,i) row each), 256 threads = 8 waves.
// Wave w owns head w (we_w B-fragments VGPR-resident). WMMA C accumulator is
// initialized with pre1T tile + row bias, so the WMMA chain directly yields
// the full pre-activation.
// ---------------------------------------------------------------------------
__global__ __launch_bounds__(256) void attn_kernel(
    const float* __restrict__ edge, const float* __restrict__ adj,
    const float* __restrict__ pre1T, const float* __restrict__ rowbias,
    const float* __restrict__ vals, const float* __restrict__ skipout,
    const float* __restrict__ a_w, const float* __restrict__ ln_scale,
    const float* __restrict__ ln_offset, const unsigned short* __restrict__ wswe,
    float* __restrict__ out){
  __shared__ __align__(16) float ebuf[2][2048];  // double-buffered 16x128 f32 tile
  __shared__ float lbuf[8*512];                  // logits/probs [head][j]
  __shared__ float obuf[128];                    // attention output
  int tid = threadIdx.x, lane = tid & 31;
  int wvs = __builtin_amdgcn_readfirstlane(tid >> 5);      // uniform head id
  int b = blockIdx.x >> 9, i = blockIdx.x & 511;
  int nl = lane & 15;
  float rb = rowbias[((size_t)(b*N_ + i))*OUT_ + wvs*16 + nl];
  float aw = a_w[wvs*16 + nl];
  // resident B fragments for this wave's head
  v16bf bfr[4];
#pragma unroll
  for (int kb = 0; kb < 4; ++kb){
    bfr[kb] = ldfrag(wswe + (wvs*4 + kb)*512 + lane*16);
  }
  const float* erow = edge + ((size_t)(b*N_ + i)) * N_ * F_;   // contiguous 256KB
  const float* p1row = pre1T + ((size_t)(b*OUT_) + wvs*16 + nl) * N_;
  int sel = (lane >> 4) * 8;

  stage_tile(erow, ebuf[0], tid);
  stage_fence();
  __syncthreads();

  for (int jt = 0; jt < 32; ++jt){
    int cur = jt & 1;
    if (jt + 1 < 32) stage_tile(erow + (jt+1)*2048, ebuf[cur ^ 1], tid);

    int jbase = jt * 16;
    // init accumulator: C = pre1[j][n] + rowbias[i][n]  (C-layout rows are
    // 8 consecutive j for this lane's half -> contiguous in pre1T)
    float4 pa = *(const float4*)(p1row + jbase + sel);
    float4 pb = *(const float4*)(p1row + jbase + sel + 4);
    v8f acc;
    acc[0] = pa.x + rb; acc[1] = pa.y + rb; acc[2] = pa.z + rb; acc[3] = pa.w + rb;
    acc[4] = pb.x + rb; acc[5] = pb.y + rb; acc[6] = pb.z + rb; acc[7] = pb.w + rb;

    const float* prow = ebuf[cur] + nl*128 + sel;
#pragma unroll
    for (int kb = 0; kb < 4; ++kb){
      float4 a0 = *(const float4*)(prow + kb*32);
      float4 a1 = *(const float4*)(prow + kb*32 + 4);
      float4 a2 = *(const float4*)(prow + kb*32 + 16);
      float4 a3 = *(const float4*)(prow + kb*32 + 20);
      v16bf afr = cat8(cvt8(a0, a1), cvt8(a2, a3));
      acc = __builtin_amdgcn_wmma_f32_16x16x32_bf16(false, afr, false, bfr[kb],
                                                    (short)0, acc, false, false);
    }
#pragma unroll
    for (int r = 0; r < 8; ++r){
      int jg = jbase + r + 8*(lane >> 4);
      float t = acc[r];
      t = t > 0.f ? t : 0.01f * t;            // leaky_relu
      float pl = t * aw;                       // dot with a_w over m (16 lanes)
      pl += __shfl_xor(pl, 1, 32);
      pl += __shfl_xor(pl, 2, 32);
      pl += __shfl_xor(pl, 4, 32);
      pl += __shfl_xor(pl, 8, 32);
      if (nl == 0) lbuf[wvs*512 + jg] = pl;
    }
    if (jt + 1 < 32) stage_fence();
    __syncthreads();
  }

  // softmax over j + weighted sum of values (wave == head; lbuf[wvs] is
  // entirely this wave's own data -> per-wave LDS ordering suffices)
  {
    int h = wvs;
    const float* arow = adj + ((size_t)(b*N_ + i))*N_;
    float lv[16]; float mx = -3.0e38f;
#pragma unroll
    for (int t = 0; t < 16; ++t){
      int j = lane + t*32;
      float l = lbuf[h*512 + j] + (arow[j] - 1.0f) * 1.0e9f;
      lv[t] = l; mx = fmaxf(mx, l);
    }
    for (int off = 16; off >= 1; off >>= 1) mx = fmaxf(mx, __shfl_xor(mx, off, 32));
    float s = 0.f;
#pragma unroll
    for (int t = 0; t < 16; ++t){
      float e = __expf(lv[t] - mx);
      s += e;
      lbuf[h*512 + lane + t*32] = e;
    }
    for (int off = 16; off >= 1; off >>= 1) s += __shfl_xor(s, off, 32);
    float inv = 1.0f / s;
    // pass 2: lane = (j-group g in 0..7) x (column quad e4)
    int e4 = (lane & 3) * 4;
    int g  = lane >> 2;
    const float* vh = vals + ((size_t)(b*H_ + h))*N_*16;
    float ax = 0.f, ay = 0.f, az = 0.f, aww = 0.f;
    for (int t = 0; t < 64; ++t){
      int j = t*8 + g;
      float p = lbuf[h*512 + j];
      float4 vv = *(const float4*)(vh + j*16 + e4);
      ax += p*vv.x; ay += p*vv.y; az += p*vv.z; aww += p*vv.w;
    }
    ax *= inv; ay *= inv; az *= inv; aww *= inv;
    for (int off = 4; off <= 16; off <<= 1){
      ax  += __shfl_xor(ax,  off, 32);
      ay  += __shfl_xor(ay,  off, 32);
      az  += __shfl_xor(az,  off, 32);
      aww += __shfl_xor(aww, off, 32);
    }
    if (g == 0){
      float4 o; o.x = ax; o.y = ay; o.z = az; o.w = aww;
      *(float4*)&obuf[h*16 + e4] = o;
    }
  }
  __syncthreads();
  if (wvs == 0){
    const float* sk = skipout + ((size_t)(b*N_ + i))*OUT_;
    float v[4]; float s = 0.f, s2 = 0.f;
#pragma unroll
    for (int q = 0; q < 4; ++q){
      int c = lane*4 + q;
      float x = obuf[c] + sk[c];
      x = fmaxf(x, 0.f);
      v[q] = x; s += x; s2 += x*x;
    }
    for (int off = 16; off >= 1; off >>= 1){
      s  += __shfl_xor(s,  off, 32);
      s2 += __shfl_xor(s2, off, 32);
    }
    float mu  = s * (1.0f/128.0f);
    float var = s2 * (1.0f/128.0f) - mu*mu;
    float rinv = rsqrtf(var + 1e-5f);
    float4 o;
    o.x = (v[0]-mu)*rinv*ln_scale[lane*4+0] + ln_offset[lane*4+0];
    o.y = (v[1]-mu)*rinv*ln_scale[lane*4+1] + ln_offset[lane*4+1];
    o.z = (v[2]-mu)*rinv*ln_scale[lane*4+2] + ln_offset[lane*4+2];
    o.w = (v[3]-mu)*rinv*ln_scale[lane*4+3] + ln_offset[lane*4+3];
    *(float4*)(out + ((size_t)(b*N_ + i))*OUT_ + lane*4) = o;
  }
}

extern "C" void kernel_launch(void* const* d_in, const int* in_sizes, int n_in,
                              void* d_out, int out_size, void* d_ws, size_t ws_size,
                              hipStream_t stream){
  const float* node     = (const float*)d_in[0];
  const float* edge     = (const float*)d_in[1];
  const float* graph    = (const float*)d_in[2];
  const float* adj      = (const float*)d_in[3];
  // d_in[4] = hidden (unused by reference)
  const float* m_w      = (const float*)d_in[5];
  const float* m_b      = (const float*)d_in[6];
  const float* skip_w   = (const float*)d_in[7];
  const float* skip_b   = (const float*)d_in[8];
  const float* w1_w     = (const float*)d_in[9];
  const float* w1_b     = (const float*)d_in[10];
  const float* w2_w     = (const float*)d_in[11];
  const float* w2_b     = (const float*)d_in[12];
  const float* we_w     = (const float*)d_in[13];
  const float* we_b     = (const float*)d_in[14];
  const float* wg_w     = (const float*)d_in[15];
  const float* wg_b     = (const float*)d_in[16];
  const float* a_w      = (const float*)d_in[17];
  // d_in[18] = a_b (constant over j -> cancels in softmax)
  const float* ln_scale = (const float*)d_in[19];
  const float* ln_offset= (const float*)d_in[20];

  char* ws = (char*)d_ws;
  unsigned short* wsw = (unsigned short*)(ws + 0);            // 5*16384 bf16 = 160 KB
  float* gbias   = (float*)(ws + 163840);
  float* vals    = (float*)(ws + 165888);                     // 1 MB
  float* pre1T   = (float*)(ws + 165888 + 1048576);           // 1 MB (transposed)
  float* rowbias = (float*)(ws + 165888 + 2*1048576);         // 1 MB
  float* skipout = (float*)(ws + 165888 + 3*1048576);         // 1 MB
  float* out = (float*)d_out;

  hipLaunchKernelGGL(prep_kernel, dim3(322), dim3(256), 0, stream,
      m_w, w1_w, w2_w, skip_w, we_w, graph, w1_b, w2_b, we_b, wg_b, wg_w, wsw, gbias);
  hipLaunchKernelGGL(proj_kernel, dim3(128), dim3(128), 0, stream,
      node, wsw, gbias, m_b, skip_b, vals, pre1T, rowbias, skipout);
  hipLaunchKernelGGL(attn_kernel, dim3(2048), dim3(256), 0, stream,
      edge, adj, pre1T, rowbias, vals, skipout, a_w, ln_scale, ln_offset,
      wsw + 4*16384, out);
}
